// Generator_19275813224466
// MI455X (gfx1250) — matrix operand order repaired and checked by reading, not compiled, not run on hardware
//
#include <hip/hip_runtime.h>
#include <hip/hip_bf16.h>

typedef __attribute__((ext_vector_type(16))) _Float16 v16h;
typedef __attribute__((ext_vector_type(8)))  _Float16 v8h;
typedef __attribute__((ext_vector_type(4)))  _Float16 v4h;
typedef __attribute__((ext_vector_type(8)))  float    v8f;

#define TPB 256
#define WAVES 8
#define ACT_STRIDE 136   // halves per row (128 + 8 pad -> conflict-free ds_load_b128)
#define ACT_HALVES (16 * ACT_STRIDE)

// misc LDS table offsets (floats)
#define OFF_B0   0
#define OFF_B1   128
#define OFF_B2   256
#define OFF_B3   320
#define OFF_V1   352
#define OFF_V2   384
#define OFF_V3   416
#define OFF_V4   448
#define OFF_V5   480
#define OFF_V6   512
#define MISC_FLOATS 544

// ---- LDS weight fragment staging -------------------------------------------
// B-fragment layout for v_wmma_f32_16x16x32_f16:
//   frag f = (kstep, ntile); lane holds 16 consecutive K for its column:
//   k = kstep*32 + (lane>=16 ? 16 : 0) + i,  n = ntile*16 + (lane&15)
// Stored lane-contiguous: frag*512 + lane*16 + i (halves) -> 32B per lane.
template<int K, int N>
__device__ inline void stageW(const float* __restrict__ Wg,
                              _Float16* __restrict__ dst, int tid) {
  constexpr int NT = N / 16;
  for (int idx = tid; idx < K * N; idx += TPB) {
    int k = idx / N;          // N is a power of two -> shift
    int n = idx - k * N;
    float w = Wg[idx];
    int ks  = k >> 5;
    int nt  = n >> 4;
    int lif = (n & 15) | (((k >> 4) & 1) << 4);
    int i   = k & 15;
    dst[((ks * NT + nt) << 9) + (lif << 4) + i] = (_Float16)w;
  }
}

__device__ inline v16h loadB(const _Float16* __restrict__ w, int lane, int frag) {
  const _Float16* p = w + (frag << 9) + (lane << 4);
  v8h lo = *(const v8h*)(p);
  v8h hi = *(const v8h*)(p + 8);
  return __builtin_shufflevector(lo, hi, 0,1,2,3,4,5,6,7,8,9,10,11,12,13,14,15);
}

// A fragment (16x32 f16): lane<16 -> row=lane, K [0..16);
//                         lane>=16 -> row=lane-16, K [16..32).
__device__ inline v16h loadA(const _Float16* __restrict__ act, int lane, int ks) {
  int r  = lane & 15;
  int hi = (lane >> 4) & 1;
  const _Float16* p = act + r * ACT_STRIDE + ks * 32 + hi * 16;
  v8h lo = *(const v8h*)(p);
  v8h hh = *(const v8h*)(p + 8);
  return __builtin_shufflevector(lo, hh, 0,1,2,3,4,5,6,7,8,9,10,11,12,13,14,15);
}

__device__ inline v8f splat8(float x) { v8f v = {x,x,x,x,x,x,x,x}; return v; }

template<int KS, int NT>
__device__ inline void mlp_layer(const _Float16* __restrict__ act,
                                 const _Float16* __restrict__ wbase,
                                 const float* __restrict__ biasLds,  // LDS
                                 v8f* c, int lane, int nlo) {
  v16h a[KS];
#pragma unroll
  for (int ks = 0; ks < KS; ++ks) a[ks] = loadA(act, lane, ks);
#pragma unroll
  for (int nt = 0; nt < NT; ++nt) c[nt] = splat8(biasLds[nt * 16 + nlo]);

  // one-fragment lookahead: B load for f+1 is in flight during wmma f
  v16h bcur = loadB(wbase, lane, 0);
#pragma unroll
  for (int f = 0; f < KS * NT; ++f) {
    v16h bnext = bcur;
    if (f + 1 < KS * NT) bnext = loadB(wbase, lane, f + 1);
    c[f % NT] = __builtin_amdgcn_wmma_f32_16x16x32_f16(
        false, a[f / NT], false, bcur, (short)0, c[f % NT], false, false);
    bcur = bnext;
  }
}

template<int NT>
__device__ inline void storeRelu(_Float16* __restrict__ act, v8f* c,
                                 int nlo, int hi8) {
#pragma unroll
  for (int nt = 0; nt < NT; ++nt) {
#pragma unroll
    for (int r = 0; r < 8; ++r) {
      act[(r + hi8) * ACT_STRIDE + nt * 16 + nlo] = (_Float16)fmaxf(c[nt][r], 0.0f);
    }
  }
}

__global__ __launch_bounds__(TPB)
void mlp_tif_wmma_kernel(const float* __restrict__ u,
                         const float* __restrict__ W0, const float* __restrict__ b0,
                         const float* __restrict__ W1, const float* __restrict__ b1,
                         const float* __restrict__ W2, const float* __restrict__ b2,
                         const float* __restrict__ W3, const float* __restrict__ b3,
                         const float* __restrict__ var1, const float* __restrict__ var2,
                         const float* __restrict__ var3, const float* __restrict__ var4,
                         const float* __restrict__ var5, const float* __restrict__ var6,
                         float* __restrict__ out, int numTiles, int totalWaves) {
  __shared__ _Float16 s_w[34816];                // 68 KB of B fragments
  __shared__ _Float16 s_act[WAVES * ACT_HALVES]; // 34 KB per-wave transpose scratch
  __shared__ float    s_misc[MISC_FLOATS];       // biases + vars (2.1 KB)

  const int tid  = threadIdx.x;
  const int lane = tid & 31;
  const int wave = tid >> 5;
  const int nlo  = lane & 15;
  const int hi8  = ((lane >> 4) & 1) << 3;

  // stage all weights (fp32 global -> f16 B fragments in LDS), once per block
  stageW< 64, 128>(W0, s_w +     0, tid);   // 16 frags
  stageW<128, 128>(W1, s_w +  8192, tid);   // 32 frags
  stageW<128,  64>(W2, s_w + 24576, tid);   // 16 frags
  stageW< 64,  32>(W3, s_w + 32768, tid);   //  4 frags

  // stage biases / vars into LDS (short-lived register reads later)
  if (tid < 128) { s_misc[OFF_B0 + tid] = b0[tid]; s_misc[OFF_B1 + tid] = b1[tid]; }
  if (tid < 64)  { s_misc[OFF_B2 + tid] = b2[tid]; }
  if (tid < 32)  {
    s_misc[OFF_B3 + tid] = b3[tid];
    s_misc[OFF_V1 + tid] = var1[tid]; s_misc[OFF_V2 + tid] = var2[tid];
    s_misc[OFF_V3 + tid] = var3[tid]; s_misc[OFF_V4 + tid] = var4[tid];
    s_misc[OFF_V5 + tid] = var5[tid]; s_misc[OFF_V6 + tid] = var6[tid];
  }
  __syncthreads();

  _Float16* act = s_act + wave * ACT_HALVES;
  const int gwave = blockIdx.x * WAVES + wave;

  for (int tile = gwave; tile < numTiles; tile += totalWaves) {
    const int rowBase = tile << 4;

    // stage 16x64 fp32 u-tile -> f16 act (coalesced b128 loads)
#pragma unroll
    for (int it = 0; it < 8; ++it) {
      int idx = it * 32 + lane;           // 256 float4 chunks
      int r   = idx >> 4;
      int c4  = (idx & 15) << 2;
      const float4 x = *(const float4*)(u + (((size_t)(rowBase + r)) << 6) + c4);
      v4h hv = { (_Float16)x.x, (_Float16)x.y, (_Float16)x.z, (_Float16)x.w };
      *(v4h*)(act + r * ACT_STRIDE + c4) = hv;
    }

    v8f c0[8], c1[8], c2[4], c3[2];
    mlp_layer<2, 8>(act, s_w +     0, s_misc + OFF_B0, c0, lane, nlo);  // 16 wmma
    storeRelu<8>(act, c0, nlo, hi8);
    mlp_layer<4, 8>(act, s_w +  8192, s_misc + OFF_B1, c1, lane, nlo);  // 32 wmma
    storeRelu<8>(act, c1, nlo, hi8);
    mlp_layer<4, 4>(act, s_w + 24576, s_misc + OFF_B2, c2, lane, nlo);  // 16 wmma
    storeRelu<4>(act, c2, nlo, hi8);
    mlp_layer<2, 2>(act, s_w + 32768, s_misc + OFF_B3, c3, lane, nlo);  //  4 wmma

    // ---- transcendental epilogue (fp32, from original u) ----
#pragma unroll
    for (int nt = 0; nt < 2; ++nt) {
      const int n = nt * 16 + nlo;        // 0..31
      const float vv1 = s_misc[OFF_V1 + n], vv2 = s_misc[OFF_V2 + n];
      const float vv3 = s_misc[OFF_V3 + n], vv4 = s_misc[OFF_V4 + n];
      const float vv5 = s_misc[OFF_V5 + n], vv6 = s_misc[OFF_V6 + n];

      float udv[8];
#pragma unroll
      for (int r = 0; r < 8; ++r)
        udv[r] = u[(((size_t)(rowBase + r + hi8)) << 6) + n];

#pragma unroll
      for (int r = 0; r < 8; ++r) {
        const int row = rowBase + r + hi8;
        const float ud  = udv[r];
        const float net = c3[nt][r];

        const float um1  = 1.0f - ud;
        const float L    = __logf(um1);
        const float invL = __builtin_amdgcn_rcpf(L);

        // Ei(L) = -E1(-L),  x = -L in (1e-4, ~9.22)
        const float x  = -L;
        const float xs = fminf(x, 8.0f);
        float s = 0.0f, term = xs;
#pragma unroll
        for (int k = 1; k <= 30; ++k) {
          s    = __builtin_fmaf(term, 1.0f / (float)k, s);
          term = term * xs * (-1.0f / (float)(k + 1));
        }
        const float e1_small = -0.5772156649015329f - __logf(xs) + s;

        const float xl  = fmaxf(x, 8.0f);
        const float rxl = __builtin_amdgcn_rcpf(xl);
        float aa = 1.0f, acc = 1.0f;
#pragma unroll
        for (int m = 1; m <= 8; ++m) { aa = aa * (-(float)m) * rxl; acc += aa; }
        const float e1_large = __expf(-xl) * rxl * acc;

        const float li = -((x < 8.0f) ? e1_small : e1_large);

        const float u2  = ud * ud;
        const float u4  = u2 * u2;
        const float p01 = u4 * __builtin_fmaf(-4.0f, ud, 5.0f);   // 5u^4 - 4u^5
        const float phi = vv2 * invL + vv3 * li
                        + vv4 * (um1 * invL - li)
                        + vv5 * (um1 * (L + 1.0f) * 0.5f * invL * invL - 0.5f * li);
        float corr1 = p01 * (vv1 + phi);
        corr1 = (corr1 != corr1) ? 0.0f : corr1;                  // nan_to_num
        corr1 = fminf(fmaxf(corr1, -3.402823466e38f), 3.402823466e38f);
        const float corr2 = vv6 * ud * um1 * um1;                 // u^3 - 2u^2 + u

        const float net_tif = fmaxf(net + corr1 + corr2, 0.0f);
        const float base = (1.0f - u2) * 0.5f;
        const float val  = exp2f(-net_tif * __log2f(base));       // base^(-net_tif)

        out[((size_t)row << 5) + n] = val;

        // keep the 16 element chains from being fully interleaved
        // (register pressure); occupancy hides the serial VALU latency
        __builtin_amdgcn_sched_barrier(0);
      }
    }
  }
}

extern "C" void kernel_launch(void* const* d_in, const int* in_sizes, int n_in,
                              void* d_out, int out_size, void* d_ws, size_t ws_size,
                              hipStream_t stream) {
  const float* u  = (const float*)d_in[0];
  const float* W0 = (const float*)d_in[1];  const float* b0 = (const float*)d_in[2];
  const float* W1 = (const float*)d_in[3];  const float* b1 = (const float*)d_in[4];
  const float* W2 = (const float*)d_in[5];  const float* b2 = (const float*)d_in[6];
  const float* W3 = (const float*)d_in[7];  const float* b3 = (const float*)d_in[8];
  const float* var1 = (const float*)d_in[9];  const float* var2 = (const float*)d_in[10];
  const float* var3 = (const float*)d_in[11]; const float* var4 = (const float*)d_in[12];
  const float* var5 = (const float*)d_in[13]; const float* var6 = (const float*)d_in[14];
  float* out = (float*)d_out;

  const int B = in_sizes[0] / 64;
  const int numTiles = B >> 4;             // 16 rows per wave-tile
  int blocks = 512;
  if (blocks * WAVES > numTiles) blocks = (numTiles + WAVES - 1) / WAVES;
  const int totalWaves = blocks * WAVES;

  mlp_tif_wmma_kernel<<<dim3(blocks), dim3(TPB), 0, stream>>>(
      u, W0, b0, W1, b1, W2, b2, W3, b3,
      var1, var2, var3, var4, var5, var6,
      out, numTiles, totalWaves);
}
